// TimeScaleAwareMamba2_24584392802586
// MI455X (gfx1250) — compile-verified, matrix operand order checked by reading, not run on hardware
//
#include <hip/hip_runtime.h>
#include <math.h>

// ---------------- problem constants ----------------
#define DM      1024
#define DS      128
#define DCONVK  4
#define HD      64
#define CK      256
#define BSZ     2
#define SEQ     4096
#define DI      2048
#define NH      32
#define CONVDIM 2304
#define DPROJ   4384
#define NC      16
#define BL      (BSZ*SEQ)      // 8192 rows total
#define EPSF    1e-5f

#define USE_ASYNC_LDS 1

typedef __attribute__((ext_vector_type(16))) __bf16 v16bf;
typedef __attribute__((ext_vector_type(8)))  float  v8f;
typedef __bf16 bf16_t;

// ---------------- helpers ----------------
__device__ __forceinline__ float softplusf_(float x) {
  return (x > 20.f) ? x : log1pf(__expf(x));
}
__device__ __forceinline__ float siluf_(float x) {
  return x / (1.f + __expf(-x));
}
__device__ __forceinline__ v8f vzero8() {
  v8f z;
#pragma unroll
  for (int i = 0; i < 8; ++i) z[i] = 0.f;
  return z;
}

__device__ __forceinline__ v8f wmma_bf16(v16bf a, v16bf b, v8f c) {
  // D = A(16x32 bf16) * B(32x16 bf16) + C(16x16 f32)
  return __builtin_amdgcn_wmma_f32_16x16x32_bf16(
      false, a, false, b, (short)0, c, false, false);
}

// Async 16B global -> LDS copy (per lane), tracked by ASYNCcnt.
__device__ __forceinline__ void async_b128(unsigned lds_off, const void* gaddr) {
  unsigned long long ga = (unsigned long long)gaddr;
  asm volatile("global_load_async_to_lds_b128 %0, %1, off"
               :: "v"(lds_off), "v"(ga) : "memory");
}
__device__ __forceinline__ void async_b128_o16(unsigned lds_off, const void* gaddr) {
  // INST_OFFSET applies to both LDS and global address -> second 16B of a 32B run
  unsigned long long ga = (unsigned long long)gaddr;
  asm volatile("global_load_async_to_lds_b128 %0, %1, off offset:16"
               :: "v"(lds_off), "v"(ga) : "memory");
}
__device__ __forceinline__ void wait_async0() {
  asm volatile("s_wait_asynccnt 0x0" ::: "memory");
}

// A-fragment: 16x32 bf16 tile from row-major storage (ld elements per row).
// lane&15 -> row; lane>>4 selects K-half; elems j<8 -> K=j, j>=8 -> K=j+8.
__device__ __forceinline__ v16bf load_frag_row(const bf16_t* src, int ld, int lane) {
  const bf16_t* p = src + (size_t)(lane & 15) * ld + (lane >> 4) * 8;
  v16bf f;
#pragma unroll
  for (int j = 0; j < 8; ++j) f[j] = p[j];
#pragma unroll
  for (int j = 0; j < 8; ++j) f[8 + j] = p[16 + j];
  return f;
}

// Swizzled fragment: 32 lanes x 16 contiguous bf16 (32B per lane slot).
__device__ __forceinline__ v16bf load_frag_sw(const bf16_t* base, int lane) {
  const bf16_t* p = base + (size_t)lane * 16;
  v16bf f;
#pragma unroll
  for (int j = 0; j < 16; ++j) f[j] = p[j];
  return f;
}

// k (0..31) -> (half, elem) of the fragment layout:
// k in {0..7}:half0 e=k ; {8..15}:half1 e=k-8 ; {16..23}:half0 e=k-8 ; {24..31}:half1 e=k-16
__device__ __forceinline__ int frag_half(int k32) { return (k32 >> 3) & 1; }
__device__ __forceinline__ int frag_elem(int k32) { return (k32 & 7) + ((k32 >> 4) << 3); }

// ---------------- kernel: f32 -> bf16 convert ----------------
__global__ void k_f32_to_bf16(const float* __restrict__ src, bf16_t* __restrict__ dst,
                              long long n) {
  long long i = (long long)blockIdx.x * blockDim.x + threadIdx.x;
  if (i < n) dst[i] = (bf16_t)src[i];
}

// ---------------- kernel: generic batched NT GEMM with WMMA ----------------
// C[M,N] = A[M,K] * B[N,K]^T ; bf16 operands, f32 out. Block tile 128x128,
// 8 waves, each wave computes a 32x64 sub-tile (2x4 WMMA tiles).
// Double-buffered LDS; tiles staged with global_load_async_to_lds_b128 so the
// copy bypasses VGPRs and overlaps with the WMMA pipe. B tile is pre-swizzled
// into the WMMA B-fragment layout (32B contiguous per lane).
__global__ __launch_bounds__(256) void k_gemm_bf16_nt(
    const bf16_t* __restrict__ A, int lda, long long sA,
    const bf16_t* __restrict__ Bm, int ldb, long long sB,
    float* __restrict__ C, int ldc, long long sC,
    int M, int N, int K) {
  __shared__ bf16_t As[2][128][48];     // [buf][m][k], 96B rows keep 16B align
  __shared__ bf16_t Bsw[2][8][32][16];  // [buf][n-group][lane][elem]
  const int bz = blockIdx.z;
  A += (long long)bz * sA;
  Bm += (long long)bz * sB;
  C += (long long)bz * sC;
  const int m0 = blockIdx.y * 128;
  const int n0 = blockIdx.x * 128;
  const int t = threadIdx.x;
  const int lane = t & 31;
  const int w = t >> 5;
  const int wr = w >> 1;               // 0..3
  const int wc = w & 1;                // 0..1

  v8f acc[2][4];
#pragma unroll
  for (int i = 0; i < 2; ++i)
#pragma unroll
    for (int j = 0; j < 4; ++j) acc[i][j] = vzero8();

  const int arow = t >> 1;             // 0..127
  const int kh = t & 1;                // which 16-wide k half
  const int acb = kh * 16;
  const int bg = arow >> 4, bln = arow & 15;
  const bool bvalid = (n0 + arow) < N;

  auto stage = [&](int buf, int k0) {
    const bf16_t* ga = A + (size_t)(m0 + arow) * lda + k0 + acb;
    bf16_t* dA0 = &As[buf][arow][acb];
    bf16_t* d0 = &Bsw[buf][bg][bln][kh * 8];
    bf16_t* d1 = &Bsw[buf][bg][bln + 16][kh * 8];
#if USE_ASYNC_LDS
    async_b128((unsigned)(size_t)dA0, ga);          // 16B
    async_b128_o16((unsigned)(size_t)dA0, ga);      // +16B (both sides)
    if (bvalid) {
      const bf16_t* gb = Bm + (size_t)(n0 + arow) * ldb + k0 + kh * 16;
      async_b128((unsigned)(size_t)d0, gb);         // k = kh*16 + 0..7
      async_b128((unsigned)(size_t)d1, gb + 8);     // k = kh*16 + 8..15
    } else {
#pragma unroll
      for (int j = 0; j < 8; ++j) d0[j] = (bf16_t)0.f;
#pragma unroll
      for (int j = 0; j < 8; ++j) d1[j] = (bf16_t)0.f;
    }
#else
#pragma unroll
    for (int j = 0; j < 16; ++j) dA0[j] = ga[j];
    if (bvalid) {
      const bf16_t* gb = Bm + (size_t)(n0 + arow) * ldb + k0 + kh * 16;
#pragma unroll
      for (int j = 0; j < 8; ++j) d0[j] = gb[j];
#pragma unroll
      for (int j = 0; j < 8; ++j) d1[j] = gb[8 + j];
    } else {
#pragma unroll
      for (int j = 0; j < 8; ++j) d0[j] = (bf16_t)0.f;
#pragma unroll
      for (int j = 0; j < 8; ++j) d1[j] = (bf16_t)0.f;
    }
#endif
  };

  stage(0, 0);
#if USE_ASYNC_LDS
  wait_async0();
#endif
  __syncthreads();

  for (int k0 = 0; k0 < K; k0 += 32) {
    const int cur = (k0 >> 5) & 1;
    if (k0 + 32 < K) stage(cur ^ 1, k0 + 32);
    v16bf af[2], bfg[4];
#pragma unroll
    for (int i = 0; i < 2; ++i)
      af[i] = load_frag_row(&As[cur][wr * 32 + i * 16][0], 48, lane);
#pragma unroll
    for (int j = 0; j < 4; ++j)
      bfg[j] = load_frag_sw(&Bsw[cur][wc * 4 + j][0][0], lane);
#pragma unroll
    for (int i = 0; i < 2; ++i)
#pragma unroll
      for (int j = 0; j < 4; ++j) acc[i][j] = wmma_bf16(af[i], bfg[j], acc[i][j]);
#if USE_ASYNC_LDS
    wait_async0();
#endif
    __syncthreads();
  }
#pragma unroll
  for (int i = 0; i < 2; ++i) {
#pragma unroll
    for (int j = 0; j < 4; ++j) {
      int col = n0 + wc * 64 + j * 16 + (lane & 15);
      if (col < N) {
        float* p = C + (size_t)(m0 + wr * 32 + i * 16 + (lane >> 4) * 8) * ldc + col;
#pragma unroll
        for (int r = 0; r < 8; ++r) p[(size_t)r * ldc] = acc[i][j][r];
      }
    }
  }
}

// ---------------- kernel: depthwise conv4 + SiLU + split ----------------
__global__ void k_conv_split(const float* __restrict__ zx, const float* __restrict__ cw,
                             const float* __restrict__ cb, const float* __restrict__ dtb,
                             const float* __restrict__ gamma, const float* __restrict__ deltat,
                             float* __restrict__ xf, bf16_t* __restrict__ Xdt,
                             bf16_t* __restrict__ Bbf, bf16_t* __restrict__ Cbf) {
  long long idx = (long long)blockIdx.x * blockDim.x + threadIdx.x;
  const long long total = (long long)BL * CONVDIM;
  if (idx >= total) return;
  int ch = (int)(idx % CONVDIM);
  long long row = idx / CONVDIM;    // b*SEQ + l
  int l = (int)(row % SEQ);
  long long rowbase = row - l;
  float acc = cb[ch];
#pragma unroll
  for (int j = 0; j < DCONVK; ++j) {
    int lj = l - (DCONVK - 1) + j;
    if (lj >= 0)
      acc += cw[ch * DCONVK + j] * zx[(rowbase + lj) * DPROJ + DI + ch];
  }
  acc = siluf_(acc);
  if (ch < DI) {
    int h = ch >> 6;  // /HD
    float dtraw = zx[row * DPROJ + DI + CONVDIM + h] + gamma[h] * deltat[row];
    float dtv = softplusf_(dtraw + dtb[h]);
    xf[row * DI + ch] = acc;
    Xdt[row * DI + ch] = (bf16_t)(acc * dtv);
  } else if (ch < DI + DS) {
    Bbf[row * DS + (ch - DI)] = (bf16_t)acc;
  } else {
    Cbf[row * DS + (ch - DI - DS)] = (bf16_t)acc;
  }
}

// ---------------- kernel: dA = dt * A ----------------
__global__ void k_compute_dA(const float* __restrict__ zx, const float* __restrict__ dtb,
                             const float* __restrict__ gamma, const float* __restrict__ deltat,
                             const float* __restrict__ Alog, float* __restrict__ dA) {
  long long idx = (long long)blockIdx.x * blockDim.x + threadIdx.x;
  if (idx >= (long long)BL * NH) return;
  int h = (int)(idx % NH);
  long long row = idx / NH;
  float dtraw = zx[row * DPROJ + DI + CONVDIM + h] + gamma[h] * deltat[row];
  float dtv = softplusf_(dtraw + dtb[h]);
  dA[idx] = -__expf(Alog[h]) * dtv;
}

// ---------------- kernel: per-(b,h,c) inclusive cumsum of dA ----------------
__global__ __launch_bounds__(256) void k_cumsum(const float* __restrict__ dA,
                                                float* __restrict__ Acs) {
  __shared__ float s[CK];
  int blk = blockIdx.x;             // ((b*NH+h)*NC + c)
  int c = blk % NC;
  int bh = blk / NC;
  int h = bh % NH;
  int b = bh / NH;
  int t = threadIdx.x;
  s[t] = dA[((long long)b * SEQ + c * CK + t) * NH + h];
  __syncthreads();
  for (int off = 1; off < CK; off <<= 1) {
    float v = (t >= off) ? s[t - off] : 0.f;
    __syncthreads();
    s[t] += v;
    __syncthreads();
  }
  Acs[(long long)blk * CK + t] = s[t];
}

// ---------------- kernel: per-(b,c,h) Y_diag (causal-masked) + chunk states ----------------
__global__ __launch_bounds__(256) void k_chunk_diag_states(
    const float* __restrict__ G,      // [B*NC][CK][CK] = C*B^T
    const bf16_t* __restrict__ Xdt,   // [BL][DI]  (x * dt, bf16)
    const bf16_t* __restrict__ Bg,    // [BL][DS]
    const float* __restrict__ Acs,    // [B][NH][NC][CK]
    float* __restrict__ Y,            // [BL][DI]
    float* __restrict__ St)           // [B*NC][NH][HD][DS]
{
  __shared__ bf16_t Xsw[8][4][32][16];   // [q-block][p-group][lane][elem] 32KB
  __shared__ bf16_t Bsw[8][8][32][16];   // [q-block][n-group][lane][elem] 64KB (decay-scaled)
  __shared__ float sAcs[CK];
  __shared__ float sDec[CK];
  const int blk = blockIdx.x;       // (b*NC + c)*NH + h
  const int h = blk % NH;
  const int zc = blk / NH;
  const int b = zc / NC;
  const int c = zc % NC;
  const int t = threadIdx.x;
  const int lane = t & 31;
  const int w = t >> 5;
  const long long lbase = (long long)zc * CK;

  sAcs[t] = Acs[(((long long)b * NH + h) * NC + c) * CK + t];
  __syncthreads();
  const float alast = sAcs[CK - 1];
  sDec[t] = __expf(alast - sAcs[t]);
  __syncthreads();

  { // stage X (swizzled) and decay-scaled B (swizzled); thread t = row q
    const int q = t;
    const int kc = q >> 5;
    const int k32 = q & 31;
    const int hf = frag_half(k32);
    const int ej = frag_elem(k32);
    const bf16_t* gx = Xdt + (lbase + q) * DI + h * HD;
#pragma unroll
    for (int n = 0; n < HD; ++n)
      Xsw[kc][n >> 4][(n & 15) + 16 * hf][ej] = gx[n];
    const float dec = sDec[q];
    const bf16_t* gb = Bg + (lbase + q) * DS;
#pragma unroll
    for (int n = 0; n < DS; ++n)
      Bsw[kc][n >> 4][(n & 15) + 16 * hf][ej] = (bf16_t)((float)gb[n] * dec);
  }
  __syncthreads();

  const float* Gc = G + (long long)zc * CK * CK;

  // ---- phase 1: Y_diag; wave w owns q in [32w, 32w+32), all 64 cols ----
  {
    const int q0 = w * 32;
    v8f acc[2][4];
#pragma unroll
    for (int i = 0; i < 2; ++i)
#pragma unroll
      for (int j = 0; j < 4; ++j) acc[i][j] = vzero8();

    auto do_block = [&](int kb, bool masked) {
      v16bf af[2];
#pragma unroll
      for (int i = 0; i < 2; ++i) {
        int m = q0 + i * 16 + (lane & 15);
        float am = sAcs[m];
        const float* gp = Gc + (size_t)m * CK + kb + (lane >> 4) * 8;
        float r0[8], r1[8];
#pragma unroll
        for (int e = 0; e < 8; ++e) r0[e] = gp[e];        // k = kb + half*8 + e
#pragma unroll
        for (int e = 0; e < 8; ++e) r1[e] = gp[16 + e];   // k = +16
        v16bf f;
#pragma unroll
        for (int j = 0; j < 16; ++j) {
          int kk = kb + ((j < 8) ? j : j + 8) + (lane >> 4) * 8;
          float g = (j < 8) ? r0[j] : r1[j - 8];
          float v = __expf(am - sAcs[kk]) * g;
          if (masked && kk > m) v = 0.f;
          f[j] = (bf16_t)v;
        }
        af[i] = f;
      }
      v16bf bfg[4];
#pragma unroll
      for (int j = 0; j < 4; ++j)
        bfg[j] = load_frag_sw(&Xsw[kb >> 5][j][0][0], lane);
#pragma unroll
      for (int i = 0; i < 2; ++i)
#pragma unroll
        for (int j = 0; j < 4; ++j) acc[i][j] = wmma_bf16(af[i], bfg[j], acc[i][j]);
    };
    for (int kb = 0; kb < q0; kb += 32) do_block(kb, false);  // strictly lower
    do_block(q0, true);                                        // diagonal block

#pragma unroll
    for (int i = 0; i < 2; ++i)
#pragma unroll
      for (int j = 0; j < 4; ++j) {
        float* p = Y + (lbase + q0 + i * 16 + (lane >> 4) * 8) * DI + h * HD +
                   j * 16 + (lane & 15);
#pragma unroll
        for (int r = 0; r < 8; ++r) p[(size_t)r * DI] = acc[i][j][r];
      }
  }
  __syncthreads();

  // ---- phase 2: chunk state (HD x DS) = X^T * (B * decay) ----
  {
    const int p0 = (w & 3) * 16;
    const int n0 = (w >> 2) * 64;
    v8f acc[4];
#pragma unroll
    for (int j = 0; j < 4; ++j) acc[j] = vzero8();
    for (int kb = 0; kb < CK; kb += 32) {
      v16bf af = load_frag_sw(&Xsw[kb >> 5][p0 >> 4][0][0], lane);
      v16bf bfg[4];
#pragma unroll
      for (int j = 0; j < 4; ++j)
        bfg[j] = load_frag_sw(&Bsw[kb >> 5][(n0 >> 4) + j][0][0], lane);
#pragma unroll
      for (int j = 0; j < 4; ++j) acc[j] = wmma_bf16(af, bfg[j], acc[j]);
    }
    float* Sb = St + ((long long)zc * NH + h) * HD * DS;
#pragma unroll
    for (int j = 0; j < 4; ++j) {
      float* p = Sb + (size_t)(p0 + (lane >> 4) * 8) * DS + n0 + j * 16 + (lane & 15);
#pragma unroll
      for (int r = 0; r < 8; ++r) p[(size_t)r * DS] = acc[j][r];
    }
  }
}

// ---------------- kernel: inter-chunk scan ----------------
__global__ __launch_bounds__(256) void k_chunk_scan(const float* __restrict__ St,
                                                    const float* __restrict__ Acs,
                                                    float* __restrict__ Pv) {
  const int b = blockIdx.x / NH, h = blockIdx.x % NH;
  const int t = threadIdx.x;
  float s[32];
#pragma unroll
  for (int i = 0; i < 32; ++i) s[i] = 0.f;
  for (int c = 0; c < NC; ++c) {
    long long base = (((long long)(b * NC + c)) * NH + h) * (long long)(HD * DS);
    float dec = __expf(Acs[(((long long)b * NH + h) * NC + c) * CK + CK - 1]);
#pragma unroll
    for (int i = 0; i < 32; ++i) {
      long long idx = base + t + i * 256;
      Pv[idx] = s[i];                 // carry BEFORE update == scan output
      s[i] = dec * s[i] + St[idx];
    }
  }
}

// ---------------- kernel: per-(b,c,h) Y_off = exp(Acs) * (C * prev^T), accumulate ----------------
__global__ __launch_bounds__(256) void k_chunk_yoff(const bf16_t* __restrict__ Cg,
                                                    const float* __restrict__ Pv,
                                                    const float* __restrict__ Acs,
                                                    float* __restrict__ Y) {
  __shared__ bf16_t Pvsw[4][4][32][16];   // [n-block(k)][p-group][lane][elem] 16KB
  __shared__ float sAcs[CK];
  const int blk = blockIdx.x;       // (b*NC + c)*NH + h
  const int h = blk % NH;
  const int zc = blk / NH;
  const int b = zc / NC;
  const int c = zc % NC;
  const int t = threadIdx.x, lane = t & 31, w = t >> 5;
  const long long lbase = (long long)zc * CK;

  { // stage prev (f32 -> bf16) swizzled: fragment is [k=n][col=p]
    const int p = t & 63;
    const int nb = (t >> 6) * 32;   // 4 threads per row, 32 n-values each
    const float* g = Pv + ((long long)zc * NH + h) * (long long)(HD * DS) +
                     (size_t)p * DS + nb;
    const int kc = nb >> 5;
    const int gp = p >> 4, lp = p & 15;
#pragma unroll
    for (int e = 0; e < 32; ++e) {
      int hf = frag_half(e);
      int ej = frag_elem(e);
      Pvsw[kc][gp][lp + 16 * hf][ej] = (bf16_t)g[e];
    }
  }
  sAcs[t] = Acs[(((long long)b * NH + h) * NC + c) * CK + t];
  __syncthreads();

  const int q0 = w * 32;
  v8f acc[2][4];
#pragma unroll
  for (int i = 0; i < 2; ++i)
#pragma unroll
    for (int j = 0; j < 4; ++j) acc[i][j] = vzero8();
  for (int kb = 0; kb < DS; kb += 32) {
    v16bf af[2];
#pragma unroll
    for (int i = 0; i < 2; ++i)
      af[i] = load_frag_row(Cg + (lbase + q0 + i * 16) * DS + kb, DS, lane);
    v16bf bfg[4];
#pragma unroll
    for (int j = 0; j < 4; ++j)
      bfg[j] = load_frag_sw(&Pvsw[kb >> 5][j][0][0], lane);
#pragma unroll
    for (int i = 0; i < 2; ++i)
#pragma unroll
      for (int j = 0; j < 4; ++j) acc[i][j] = wmma_bf16(af[i], bfg[j], acc[i][j]);
  }
#pragma unroll
  for (int i = 0; i < 2; ++i)
#pragma unroll
    for (int j = 0; j < 4; ++j)
#pragma unroll
      for (int r = 0; r < 8; ++r) {
        int q = q0 + i * 16 + (lane >> 4) * 8 + r;
        float sc = __expf(sAcs[q]);
        float* p = Y + (lbase + q) * DI + h * HD + j * 16 + (lane & 15);
        *p += sc * acc[i][j][r];
      }
}

// ---------------- kernel: gate (silu) + RMSNorm -> bf16 activations ----------------
__global__ __launch_bounds__(256) void k_gate_norm(const float* __restrict__ Y,
                                                   const float* __restrict__ xf,
                                                   const float* __restrict__ zx,
                                                   const float* __restrict__ Dp,
                                                   const float* __restrict__ nw,
                                                   bf16_t* __restrict__ Yn) {
  __shared__ float red[256];
  const long long row = blockIdx.x;
  const int t = threadIdx.x;
  float vals[8];
  float ss = 0.f;
#pragma unroll
  for (int i = 0; i < 8; ++i) {
    int ch = t + i * 256;
    float y = Y[row * DI + ch] + xf[row * DI + ch] * Dp[ch >> 6];
    float z = zx[row * DPROJ + ch];
    float g = y * siluf_(z);
    vals[i] = g;
    ss += g * g;
  }
  red[t] = ss;
  __syncthreads();
  for (int s = 128; s > 0; s >>= 1) {
    if (t < s) red[t] += red[t + s];
    __syncthreads();
  }
  float r = rsqrtf(red[0] / (float)DI + EPSF);
#pragma unroll
  for (int i = 0; i < 8; ++i) {
    int ch = t + i * 256;
    Yn[row * DI + ch] = (bf16_t)(vals[i] * r * nw[ch]);
  }
}

// ---------------- host launch ----------------
extern "C" void kernel_launch(void* const* d_in, const int* in_sizes, int n_in,
                              void* d_out, int out_size, void* d_ws, size_t ws_size,
                              hipStream_t stream) {
  const float* u       = (const float*)d_in[0];
  const float* deltat  = (const float*)d_in[1];
  const float* W_in    = (const float*)d_in[2];
  const float* conv_w  = (const float*)d_in[3];
  const float* conv_b  = (const float*)d_in[4];
  const float* dt_bias = (const float*)d_in[5];
  const float* gamma   = (const float*)d_in[6];
  const float* A_log   = (const float*)d_in[7];
  const float* Dp      = (const float*)d_in[8];
  const float* norm_w  = (const float*)d_in[9];
  const float* W_out   = (const float*)d_in[10];
  float* out = (float*)d_out;

  char* wp = (char*)d_ws;
  auto alloc = [&](size_t bytes) -> void* {
    void* p = (void*)wp;
    wp += (bytes + 255) & ~(size_t)255;
    return p;
  };
  bf16_t* u_bf  = (bf16_t*)alloc((size_t)BL * DM * 2);
  bf16_t* Wi_bf = (bf16_t*)alloc((size_t)DPROJ * DM * 2);
  bf16_t* Wo_bf = (bf16_t*)alloc((size_t)DM * DI * 2);
  float*  zx    = (float*)alloc((size_t)BL * DPROJ * 4);
  float*  xf    = (float*)alloc((size_t)BL * DI * 4);
  bf16_t* Xdt   = (bf16_t*)alloc((size_t)BL * DI * 2);
  bf16_t* Bbf   = (bf16_t*)alloc((size_t)BL * DS * 2);
  bf16_t* Cbf   = (bf16_t*)alloc((size_t)BL * DS * 2);
  float*  dA    = (float*)alloc((size_t)BL * NH * 4);
  float*  Acs   = (float*)alloc((size_t)BSZ * NH * NC * CK * 4);
  float*  G     = (float*)alloc((size_t)BSZ * NC * CK * CK * 4);
  float*  St    = (float*)alloc((size_t)BSZ * NC * NH * HD * DS * 4);
  float*  Pv    = (float*)alloc((size_t)BSZ * NC * NH * HD * DS * 4);
  float*  Yb    = (float*)alloc((size_t)BL * DI * 4);
  bf16_t* Yn    = (bf16_t*)alloc((size_t)BL * DI * 2);

  auto cvt = [&](const float* s, bf16_t* d, long long n) {
    int g = (int)((n + 255) / 256);
    k_f32_to_bf16<<<g, 256, 0, stream>>>(s, d, n);
  };
  cvt(u, u_bf, (long long)BL * DM);
  cvt(W_in, Wi_bf, (long long)DPROJ * DM);
  cvt(W_out, Wo_bf, (long long)DM * DI);

  // 1) in-projection: zx[BL][DPROJ] = u @ W_in^T
  {
    dim3 grid((DPROJ + 127) / 128, BL / 128, 1);
    k_gemm_bf16_nt<<<grid, 256, 0, stream>>>(u_bf, DM, 0, Wi_bf, DM, 0,
                                             zx, DPROJ, 0, BL, DPROJ, DM);
  }
  // 2) conv + silu + split, dt fold-in
  {
    long long total = (long long)BL * CONVDIM;
    k_conv_split<<<(int)((total + 255) / 256), 256, 0, stream>>>(
        zx, conv_w, conv_b, dt_bias, gamma, deltat, xf, Xdt, Bbf, Cbf);
  }
  // 3) dA and per-chunk cumsum
  {
    long long total = (long long)BL * NH;
    k_compute_dA<<<(int)((total + 255) / 256), 256, 0, stream>>>(
        zx, dt_bias, gamma, deltat, A_log, dA);
    k_cumsum<<<BSZ * NH * NC, 256, 0, stream>>>(dA, Acs);
  }
  // 4) G = C @ B^T per chunk (batched over b*c)
  {
    dim3 grid(CK / 128, CK / 128, BSZ * NC);
    k_gemm_bf16_nt<<<grid, 256, 0, stream>>>(
        Cbf, DS, (long long)CK * DS, Bbf, DS, (long long)CK * DS,
        G, CK, (long long)CK * CK, CK, CK, DS);
  }
  // 5) Y_diag + chunk states
  k_chunk_diag_states<<<BSZ * NC * NH, 256, 0, stream>>>(G, Xdt, Bbf, Acs, Yb, St);
  // 6) inter-chunk scan
  k_chunk_scan<<<BSZ * NH, 256, 0, stream>>>(St, Acs, Pv);
  // 7) Y_off accumulate
  k_chunk_yoff<<<BSZ * NC * NH, 256, 0, stream>>>(Cbf, Pv, Acs, Yb);
  // 8) gate + RMSNorm
  k_gate_norm<<<BL, 256, 0, stream>>>(Yb, xf, zx, Dp, norm_w, Yn);
  // 9) out-projection: out[BL][DM] = Yn @ W_out^T
  {
    dim3 grid(DM / 128, BL / 128, 1);
    k_gemm_bf16_nt<<<grid, 256, 0, stream>>>(Yn, DI, 0, Wo_bf, DI, 0,
                                             out, DM, 0, BL, DM, DI);
  }
}
